// TransformerLayer_40089224741279
// MI455X (gfx1250) — matrix operand order repaired and checked by reading, not compiled, not run on hardware
//
#include <hip/hip_runtime.h>
#include <hip/hip_bf16.h>

typedef __attribute__((ext_vector_type(16))) _Float16 v16h;
typedef __attribute__((ext_vector_type(8)))  _Float16 v8h;
typedef __attribute__((ext_vector_type(8)))  float    v8f;

#define BB 2
#define SS 2048
#define DD 1024
#define HH 16
#define DH 64
#define MM (BB * SS)   // 4096
#define FF (2 * DD)    // 2048

__device__ __forceinline__ v16h join16(v8h lo, v8h hi) {
  return __builtin_shufflevector(lo, hi, 0,1,2,3,4,5,6,7,8,9,10,11,12,13,14,15);
}
__device__ __forceinline__ v8h ldg8h(const _Float16* p) {
  return *reinterpret_cast<const v8h*>(p);
}

// A fragment, 16x32 f16, row-major A with leading dim lda.
// lane l (<16): row = rowBase + l, halves = k in {kb..kb+7, kb+16..kb+23}
// lane l (>=16): same row, k in {kb+8..kb+15, kb+24..kb+31}
__device__ __forceinline__ v16h load_a16x32(const _Float16* A, int lda,
                                            int rowBase, int kb, int lane) {
  int m  = rowBase + (lane & 15);
  int ko = kb + ((lane & 16) ? 8 : 0);
  const _Float16* p = A + (size_t)m * lda + ko;
  return join16(ldg8h(p), ldg8h(p + 16));
}

// B fragment, 32x16 f16, from "transposed" storage Bt (row n = contiguous K).
// lane l: col n = nBase + (l&15), halves = k in {kb+8h..+7, kb+16+8h..+7}, h = l>=16.
__device__ __forceinline__ v16h load_bt32x16(const _Float16* Bt, int ldb,
                                             int nBase, int kb, int lane) {
  int n  = nBase + (lane & 15);
  int ko = kb + ((lane & 16) ? 8 : 0);
  const _Float16* p = Bt + (size_t)n * ldb + ko;
  return join16(ldg8h(p), ldg8h(p + 16));
}

__device__ __forceinline__ v8f wmma_f16(v16h a, v16h b, v8f c) {
  return __builtin_amdgcn_wmma_f32_16x16x32_f16(false, a, false, b,
                                                (short)0, c, false, false);
}

// Async global->LDS copy of one 16-byte chunk (CDNA5, tracked by ASYNCcnt).
__device__ __forceinline__ void async_b128(unsigned int lds_byte_addr,
                                           const void* gptr) {
  unsigned long long ga = (unsigned long long)(size_t)gptr;
  asm volatile("global_load_async_to_lds_b128 %0, %1, off"
               :: "v"(lds_byte_addr), "v"(ga) : "memory");
}
__device__ __forceinline__ void wait_async0() {
  asm volatile("s_wait_asynccnt 0" ::: "memory");
}

// ---------------------------------------------------------------- converts
__global__ __launch_bounds__(256) void k_cvt_f16(const float* __restrict__ in,
                                                 _Float16* __restrict__ out,
                                                 int n) {
  int i = blockIdx.x * 256 + threadIdx.x;
  if (i < n) out[i] = (_Float16)in[i];
}

// Wt[n*K + k] = (f16) W[k*N + n]   (tid: k fastest)
__global__ __launch_bounds__(256) void k_tr_cvt(const float* __restrict__ W,
                                                _Float16* __restrict__ Wt,
                                                int K, int N) {
  size_t tid = (size_t)blockIdx.x * 256 + threadIdx.x;
  int k = (int)(tid % K);
  int n = (int)(tid / K);
  Wt[tid] = (_Float16)W[(size_t)k * N + n];
}

// Vt[((b*H+h)*DH + d)*S + s] = Vh[(b*S+s)*D + h*DH + d]   (tid: s fastest)
__global__ __launch_bounds__(256) void k_tr_v(const _Float16* __restrict__ Vh,
                                              _Float16* __restrict__ Vt) {
  size_t tid = (size_t)blockIdx.x * 256 + threadIdx.x;
  int s = (int)(tid & (SS - 1));
  size_t rest = tid >> 11;
  int d = (int)(rest & (DH - 1));
  int h = (int)((rest >> 6) & (HH - 1));
  int b = (int)(rest >> 10);
  Vt[tid] = Vh[((size_t)b * SS + s) * DD + h * DH + d];
}

__global__ __launch_bounds__(256) void k_mask(const int* __restrict__ m,
                                              float* __restrict__ madd, int n) {
  int i = blockIdx.x * 256 + threadIdx.x;
  if (i < n) madd[i] = m[i] ? 0.0f : -1e9f;
}

// ---------------------------------------------------------------- GEMM
// C[M,N] = A[M,K](f16) * Bt[N,K]^T(f16) + bias (+residual)(+relu)
// Block = 256 threads = 8 waves; macro tile 128(M) x 64(N); wave w owns rows
// [w*16, w*16+16). The 64x32 B panel per k-step is staged in double-buffered
// LDS via async global->LDS copies shared by all 8 waves.
__global__ __launch_bounds__(256) void k_gemm(const _Float16* __restrict__ A,
                                              const _Float16* __restrict__ Bt,
                                              const float* __restrict__ bias,
                                              const float* __restrict__ residual,
                                              float* __restrict__ outF,
                                              _Float16* __restrict__ outH,
                                              int M, int N, int K, int relu) {
  __shared__ _Float16 Bs[2][64 * 32];   // 2 x 4KB panels
  int t    = threadIdx.x;
  int wv   = t >> 5;
  int lane = t & 31;
  int tilesM = M >> 7;                  // M/128
  int mt = blockIdx.x % tilesM;
  int nt = blockIdx.x / tilesM;
  int rowBase = mt * 128 + wv * 16;
  int colBase = nt * 64;

  // per-thread slice of the B panel: n = t/4, k-offset = (t%4)*8 halves (16B)
  int pn = t >> 2;
  int pk = (t & 3) * 8;
  unsigned int ldsBase0 = (unsigned int)(size_t)&Bs[0][pn * 32 + pk];
  unsigned int ldsBase1 = (unsigned int)(size_t)&Bs[1][pn * 32 + pk];
  const _Float16* gB = Bt + (size_t)(colBase + pn) * K + pk;

  async_b128(ldsBase0, gB);             // prefetch k-step 0 into buf 0

  v8f acc[4] = {v8f{}, v8f{}, v8f{}, v8f{}};
  int nsteps = K >> 5;
  for (int s = 0; s < nsteps; ++s) {
    int kb = s << 5;
    wait_async0();
    __syncthreads();                    // panel for step s is ready for all
    if (s + 1 < nsteps)                 // overlap next panel with compute
      async_b128((s & 1) ? ldsBase0 : ldsBase1, gB + kb + 32);

    v16h a = load_a16x32(A, K, rowBase, kb, lane);
    const _Float16* Bb = &Bs[s & 1][0];
    int ko = (lane & 16) ? 8 : 0;
#pragma unroll
    for (int j = 0; j < 4; ++j) {
      int n2 = j * 16 + (lane & 15);
      const v8h* p = reinterpret_cast<const v8h*>(Bb + n2 * 32 + ko);
      v16h b = join16(p[0], p[2]);
      acc[j] = wmma_f16(a, b, acc[j]);
    }
  }

  int rRow = rowBase + ((lane & 16) ? 8 : 0);
  int col0 = colBase + (lane & 15);
#pragma unroll
  for (int j = 0; j < 4; ++j) {
    int col = col0 + j * 16;
    float bv = bias ? bias[col] : 0.0f;
#pragma unroll
    for (int r = 0; r < 8; ++r) {
      int row = rRow + r;
      float v = acc[j][r] + bv;
      if (residual) v += residual[(size_t)row * N + col];
      if (relu) v = v > 0.0f ? v : 0.0f;
      if (outF) outF[(size_t)row * N + col] = v;
      if (outH) outH[(size_t)row * N + col] = (_Float16)v;
    }
  }
}

// ---------------------------------------------------------------- attention
// One wave handles (b,h) and 16 query rows; flash-style over 32-key steps.
__global__ __launch_bounds__(128) void k_attn(const _Float16* __restrict__ Qh,
                                              const _Float16* __restrict__ Kh,
                                              const _Float16* __restrict__ Vt,
                                              const float* __restrict__ maskadd,
                                              _Float16* __restrict__ attnH) {
  __shared__ float pbuf[4][16 * 32];
  int wslot = threadIdx.x >> 5;
  int lane  = threadIdx.x & 31;
  int wid = blockIdx.x * 4 + wslot;
  int qt = wid & 127;            // S/16 = 128 query tiles
  int h  = (wid >> 7) & 15;
  int b  = wid >> 11;

  const _Float16* Qbase = Qh + ((size_t)(b * SS + qt * 16)) * DD + h * DH;
  const _Float16* Kbase = Kh + (size_t)b * SS * DD + h * DH;
  const _Float16* Vtb   = Vt + ((size_t)(b * HH + h) * DH) * SS;
  const float*    mrowp = maskadd + (size_t)b * SS;

  v16h qa0 = load_a16x32(Qbase, DD, 0, 0, lane);
  v16h qa1 = load_a16x32(Qbase, DD, 0, 32, lane);

  float mrow[8], srow[8];
#pragma unroll
  for (int r = 0; r < 8; ++r) { mrow[r] = -3.0e38f; srow[r] = 0.0f; }
  v8f o[4] = {v8f{}, v8f{}, v8f{}, v8f{}};
  float* P = pbuf[wslot];

  for (int kt = 0; kt < SS / 32; ++kt) {
    int key0 = kt * 32;
    v8f s0 = {}, s1 = {};
    {
      v16h kb0 = load_bt32x16(Kbase, DD, key0, 0, lane);
      v16h kb1 = load_bt32x16(Kbase, DD, key0, 32, lane);
      s0 = wmma_f16(qa0, kb0, s0);
      s0 = wmma_f16(qa1, kb1, s0);
      v16h kc0 = load_bt32x16(Kbase, DD, key0 + 16, 0, lane);
      v16h kc1 = load_bt32x16(Kbase, DD, key0 + 16, 32, lane);
      s1 = wmma_f16(qa0, kc0, s1);
      s1 = wmma_f16(qa1, kc1, s1);
    }
    float madd0 = mrowp[key0 + (lane & 15)];
    float madd1 = mrowp[key0 + 16 + (lane & 15)];
#pragma unroll
    for (int r = 0; r < 8; ++r) {
      s0[r] = s0[r] * 0.125f + madd0;   // 1/sqrt(64)
      s1[r] = s1[r] * 0.125f + madd1;
    }
    float alpha[8];
#pragma unroll
    for (int r = 0; r < 8; ++r) {
      float mt_ = fmaxf(s0[r], s1[r]);
#pragma unroll
      for (int off = 8; off >= 1; off >>= 1)
        mt_ = fmaxf(mt_, __shfl_xor(mt_, off, 32));
      float mn = fmaxf(mrow[r], mt_);
      float al = __expf(mrow[r] - mn);
      float p0 = __expf(s0[r] - mn);
      float p1 = __expf(s1[r] - mn);
      s0[r] = p0; s1[r] = p1;
      float tt = p0 + p1;
#pragma unroll
      for (int off = 8; off >= 1; off >>= 1)
        tt += __shfl_xor(tt, off, 32);
      srow[r] = srow[r] * al + tt;
      mrow[r] = mn;
      alpha[r] = al;
    }
#pragma unroll
    for (int j = 0; j < 4; ++j)
#pragma unroll
      for (int r = 0; r < 8; ++r) o[j][r] *= alpha[r];

    // probs (C layout) -> LDS -> A-fragment layout (wave-private slice)
    {
      int prow = (lane & 16) ? 8 : 0;
      int pcol = lane & 15;
#pragma unroll
      for (int r = 0; r < 8; ++r) {
        P[(prow + r) * 32 + pcol]      = s0[r];
        P[(prow + r) * 32 + 16 + pcol] = s1[r];
      }
    }
    v16h pa;
    {
      const float* Pr = &P[(lane & 15) * 32];
      int b0 = (lane & 16) ? 8 : 0;
#pragma unroll
      for (int i = 0; i < 8; ++i) pa[i] = (_Float16)Pr[b0 + i];
#pragma unroll
      for (int i = 0; i < 8; ++i) pa[8 + i] = (_Float16)Pr[16 + b0 + i];
    }
#pragma unroll
    for (int j = 0; j < 4; ++j) {
      v16h vb = load_bt32x16(Vtb, SS, j * 16, key0, lane);
      o[j] = wmma_f16(pa, vb, o[j]);
    }
  }

  int rowBase = qt * 16 + ((lane & 16) ? 8 : 0);
#pragma unroll
  for (int j = 0; j < 4; ++j) {
#pragma unroll
    for (int r = 0; r < 8; ++r) {
      float v = o[j][r] / srow[r];
      size_t row = (size_t)b * SS + rowBase + r;
      attnH[row * DD + h * DH + j * 16 + (lane & 15)] = (_Float16)v;
    }
  }
}

// ---------------------------------------------------------------- layernorm
__global__ __launch_bounds__(256) void k_ln(const float* __restrict__ in,
                                            const float* __restrict__ g,
                                            const float* __restrict__ be,
                                            float* __restrict__ outF,
                                            _Float16* __restrict__ outH) {
  int row = blockIdx.x;
  const float* x = in + (size_t)row * DD;
  float v[4], s = 0.0f, ss = 0.0f;
#pragma unroll
  for (int i = 0; i < 4; ++i) {
    v[i] = x[threadIdx.x + i * 256];
    s += v[i];
    ss += v[i] * v[i];
  }
#pragma unroll
  for (int off = 16; off >= 1; off >>= 1) {
    s  += __shfl_xor(s, off, 32);
    ss += __shfl_xor(ss, off, 32);
  }
  __shared__ float sh[2][8];
  int w = threadIdx.x >> 5, l = threadIdx.x & 31;
  if (l == 0) { sh[0][w] = s; sh[1][w] = ss; }
  __syncthreads();
  s = 0.0f; ss = 0.0f;
#pragma unroll
  for (int i = 0; i < 8; ++i) { s += sh[0][i]; ss += sh[1][i]; }
  float mean = s * (1.0f / DD);
  float var  = ss * (1.0f / DD) - mean * mean;
  float inv  = rsqrtf(var + 1e-5f);
#pragma unroll
  for (int i = 0; i < 4; ++i) {
    int c = threadIdx.x + i * 256;
    float y = (v[i] - mean) * inv * g[c] + be[c];
    if (outF) outF[(size_t)row * DD + c] = y;
    if (outH) outH[(size_t)row * DD + c] = (_Float16)y;
  }
}

// ---------------------------------------------------------------- launch
extern "C" void kernel_launch(void* const* d_in, const int* in_sizes, int n_in,
                              void* d_out, int out_size, void* d_ws, size_t ws_size,
                              hipStream_t stream) {
  (void)in_sizes; (void)n_in; (void)out_size; (void)ws_size;
  const float* x    = (const float*)d_in[0];
  const int*   mask = (const int*)d_in[1];
  const float* wq = (const float*)d_in[2];  const float* bq = (const float*)d_in[3];
  const float* wk = (const float*)d_in[4];  const float* bk = (const float*)d_in[5];
  const float* wv = (const float*)d_in[6];  const float* bv = (const float*)d_in[7];
  const float* wo = (const float*)d_in[8];  const float* bo = (const float*)d_in[9];
  const float* ln1g = (const float*)d_in[10]; const float* ln1b = (const float*)d_in[11];
  const float* w1 = (const float*)d_in[12]; const float* b1 = (const float*)d_in[13];
  const float* w2 = (const float*)d_in[14]; const float* b2 = (const float*)d_in[15];
  const float* ln2g = (const float*)d_in[16]; const float* ln2b = (const float*)d_in[17];
  float* out = (float*)d_out;

  char* cur = (char*)d_ws;
  auto take = [&](size_t bytes) {
    void* p = cur;
    cur += (bytes + 255) & ~size_t(255);
    return p;
  };
  _Float16* Xh   = (_Float16*)take((size_t)MM * DD * 2);
  _Float16* WqT  = (_Float16*)take((size_t)DD * DD * 2);
  _Float16* WkT  = (_Float16*)take((size_t)DD * DD * 2);
  _Float16* WvT  = (_Float16*)take((size_t)DD * DD * 2);
  _Float16* WoT  = (_Float16*)take((size_t)DD * DD * 2);
  _Float16* W1T  = (_Float16*)take((size_t)FF * DD * 2);
  _Float16* W2T  = (_Float16*)take((size_t)DD * FF * 2);
  _Float16* Qh   = (_Float16*)take((size_t)MM * DD * 2);
  _Float16* Kh   = (_Float16*)take((size_t)MM * DD * 2);
  _Float16* Vh   = (_Float16*)take((size_t)MM * DD * 2);
  _Float16* Vt   = (_Float16*)take((size_t)MM * DD * 2);
  float*    madd = (float*)take((size_t)MM * 4);
  _Float16* Ah   = (_Float16*)take((size_t)MM * DD * 2);
  float*    aOut = (float*)take((size_t)MM * DD * 4);
  float*    hF   = (float*)take((size_t)MM * DD * 4);
  _Float16* hH   = (_Float16*)take((size_t)MM * DD * 2);
  _Float16* ff1  = (_Float16*)take((size_t)MM * FF * 2);
  float*    ffo  = (float*)take((size_t)MM * DD * 4);

  // converts
  k_cvt_f16<<<(MM * DD) / 256, 256, 0, stream>>>(x, Xh, MM * DD);
  k_mask<<<MM / 256, 256, 0, stream>>>(mask, madd, MM);
  k_tr_cvt<<<(DD * DD) / 256, 256, 0, stream>>>(wq, WqT, DD, DD);
  k_tr_cvt<<<(DD * DD) / 256, 256, 0, stream>>>(wk, WkT, DD, DD);
  k_tr_cvt<<<(DD * DD) / 256, 256, 0, stream>>>(wv, WvT, DD, DD);
  k_tr_cvt<<<(DD * DD) / 256, 256, 0, stream>>>(wo, WoT, DD, DD);
  k_tr_cvt<<<(DD * FF) / 256, 256, 0, stream>>>(w1, W1T, DD, FF);  // Wt[FF,DD]
  k_tr_cvt<<<(FF * DD) / 256, 256, 0, stream>>>(w2, W2T, FF, DD);  // Wt[DD,FF]

  // Q, K, V projections (f16 outputs only): grid = (M/128)*(N/64) blocks
  int gQ = (MM / 128) * (DD / 64);   // 512
  k_gemm<<<gQ, 256, 0, stream>>>(Xh, WqT, bq, nullptr, nullptr, Qh, MM, DD, DD, 0);
  k_gemm<<<gQ, 256, 0, stream>>>(Xh, WkT, bk, nullptr, nullptr, Kh, MM, DD, DD, 0);
  k_gemm<<<gQ, 256, 0, stream>>>(Xh, WvT, bv, nullptr, nullptr, Vh, MM, DD, DD, 0);
  k_tr_v<<<(MM * DD) / 256, 256, 0, stream>>>(Vh, Vt);

  // attention: B*H*(S/16) = 4096 waves
  k_attn<<<(BB * HH * (SS / 16)) / 4, 128, 0, stream>>>(Qh, Kh, Vt, madd, Ah);

  // out-proj + residual x
  k_gemm<<<gQ, 256, 0, stream>>>(Ah, WoT, bo, x, aOut, nullptr, MM, DD, DD, 0);
  k_ln<<<MM, 256, 0, stream>>>(aOut, ln1g, ln1b, hF, hH);

  // FFN
  int gF1 = (MM / 128) * (FF / 64);  // 1024
  k_gemm<<<gF1, 256, 0, stream>>>(hH, W1T, b1, nullptr, nullptr, ff1, MM, FF, DD, 1);
  k_gemm<<<gQ, 256, 0, stream>>>(ff1, W2T, b2, hF, ffo, nullptr, MM, DD, FF, 0);
  k_ln<<<MM, 256, 0, stream>>>(ffo, ln2g, ln2b, out, nullptr);
}
